// GCPNetUpdate_68401649156276
// MI455X (gfx1250) — matrix operand order repaired and checked by reading, not compiled, not run on hardware
//
#include <hip/hip_runtime.h>
#include <hip/hip_bf16.h>

// ---------------------------------------------------------------------------
// GCPNet update (two GCAO layers) for MI455X / gfx1250, wave32 + WMMA f16.
// - Fragment-major packing: every WMMA operand fragment is a contiguous
//   32-byte per-lane block (2x b128 loads).
// - hi/hj share B fragments in a single k-loop (2 WMMAs per B fetch).
// - hj persisted in C-fragment-major layout: one b128 store per tile.
// ---------------------------------------------------------------------------

typedef _Float16 f16;
typedef __attribute__((ext_vector_type(16))) _Float16 v16h;
typedef __attribute__((ext_vector_type(8)))  _Float16 v8h;
typedef __attribute__((ext_vector_type(8)))  float    v8f;

#define DIM 128
#define NH  4

#define WMMA_F32_F16(a, b, c) \
  __builtin_amdgcn_wmma_f32_16x16x32_f16(false, (a), false, (b), (short)0, (c), false, false)

__device__ __forceinline__ float sigmoidf_(float x) {
  return __builtin_amdgcn_rcpf(1.0f + __expf(-x));
}
__device__ __forceinline__ float siluf_(float x)    { return x * sigmoidf_(x); }
__device__ __forceinline__ float softplusf_(float x){
  return (x > 20.0f) ? x : __logf(1.0f + __expf(x));
}

// Order-preserving float->uint encoding for atomicMax-based segment max.
// memset(0) init decodes to NaN -> treated as "empty segment" -> clamped to 0.
__device__ __forceinline__ unsigned encf_(float x) {
  unsigned u = __float_as_uint(x);
  return (u & 0x80000000u) ? ~u : (u | 0x80000000u);
}
__device__ __forceinline__ float decf_(unsigned u) {
  return (u & 0x80000000u) ? __uint_as_float(u ^ 0x80000000u) : __uint_as_float(~u);
}

// Fragment-major A tile in LDS: [kstep][lane][16 elems] f16.
// ISA A-layout forward map (lane,e) -> k within 32-wide kstep:
//   kl = ((e&8)?16:0) + ((lane&16)?8:0) + (e&7),  row = lane&15
__device__ __forceinline__ int a_klocal_(int lane, int e) {
  return ((e & 8) ? 16 : 0) + ((lane & 16) ? 8 : 0) + (e & 7);
}

__device__ __forceinline__ v16h fragA(const f16* Ap, int ks, int lane) {
  return *(const v16h*)(Ap + (((ks << 5) + lane) << 4));
}
// Packed B: [tile = kt*NT+nt][lane][16 elems]
__device__ __forceinline__ v16h fragB(const f16* Bp, int tile, int lane) {
  return *(const v16h*)(Bp + ((((size_t)tile << 5) + lane) << 4));
}

// ---------------------------------------------------------------------------
// Pack [K][N] f32 row-major weights into fragment-major f16.
// ISA B-layout: col = lane&15, kl = ((e&8)?16:0) + ((lane&16)?8:0) + (e&7)
__global__ void k_pack_B(const float* __restrict__ src, f16* __restrict__ dst, int K, int N) {
  int i = blockIdx.x * blockDim.x + threadIdx.x;  // destination index
  if (i >= K * N) return;
  int e    = i & 15;
  int lane = (i >> 4) & 31;
  int tile = i >> 9;
  int NT   = N >> 4;
  int kt   = tile / NT, nt = tile - kt * NT;
  int col  = nt * 16 + (lane & 15);
  int kk   = kt * 32 + ((e & 8) ? 16 : 0) + ((lane & 16) ? 8 : 0) + (e & 7);
  dst[i] = (f16)src[(size_t)kk * N + col];
}

// ---------------------------------------------------------------------------
// K1: per 16-edge block: hi/hj = silu([x[idx]|ea] @ W)  (16x256 @ 256x512 WMMA)
// stores hj (f16, C-fragment-major) and alpha_raw = silu(sum_d hi*att_a + hj*att_b)
__global__ __launch_bounds__(128) void k_gemm_h_alpha(
    const float* __restrict__ x, const float* __restrict__ ea,
    const int* __restrict__ eidx, int E,
    const f16* __restrict__ Wp, const float* __restrict__ att,
    f16* __restrict__ hj_out, float* __restrict__ alpha_raw) {
  __shared__ __align__(32) f16 Ai[16 * 256];   // fragment-major: 8 ksteps x 32 lanes x 16
  __shared__ __align__(32) f16 Aj[16 * 256];
  __shared__ float AttS[NH * 256];
  __shared__ float AlphaAcc[16 * NH];
  __shared__ int   si[16], sj[16];

  int tid = threadIdx.x;
  int e0  = blockIdx.x * 16;
  if (tid < 16) { si[tid] = eidx[e0 + tid]; sj[tid] = eidx[E + e0 + tid]; }
  if (tid < 16 * NH) AlphaAcc[tid] = 0.0f;
  for (int k = tid; k < NH * 256; k += 128) AttS[k] = att[k];
  __syncthreads();
  // Stage A tiles in fragment order: iterate destination, compute source elem.
  for (int di = tid; di < 16 * 256; di += 128) {
    int e    = di & 15;
    int ln   = (di >> 4) & 31;
    int ks   = di >> 9;
    int r    = ln & 15;
    int c    = ks * 32 + a_klocal_(ln, e);
    float vi = (c < DIM) ? x[(size_t)si[r] * DIM + c] : ea[(size_t)(e0 + r) * DIM + (c - DIM)];
    float vj = (c < DIM) ? x[(size_t)sj[r] * DIM + c] : ea[(size_t)(e0 + r) * DIM + (c - DIM)];
    Ai[di] = (f16)vi;
    Aj[di] = (f16)vj;
  }
  __syncthreads();

  int wave = tid >> 5, lane = tid & 31;
  int rowc = (lane & 16) ? 8 : 0;
  int colc = lane & 15;

#pragma unroll 1
  for (int t8 = 0; t8 < 8; ++t8) {
    int tile = wave + t8 * 4;          // 0..31 -> 512 output cols, NT = 32
    int n0   = tile * 16;
    int h    = n0 >> 7;                // head index (128 cols per head)
    int cwi  = (n0 & 127) + colc;      // col within head

    // shared B fragments: 2 WMMAs per fetch
    v8f ci = {}, cj = {};
#pragma unroll
    for (int ks = 0; ks < 8; ++ks) {
      v16h b = fragB(Wp, ks * 32 + tile, lane);
      ci = WMMA_F32_F16(fragA(Ai, ks, lane), b, ci);
      cj = WMMA_F32_F16(fragA(Aj, ks, lane), b, cj);
    }

    v8h hjv;
#pragma unroll
    for (int r = 0; r < 8; ++r) {
      int   m  = r + rowc;
      float vi = siluf_(ci[r]);
      float vj = siluf_(cj[r]);
      atomicAdd(&AlphaAcc[m * NH + h],
                vi * AttS[h * 256 + cwi] + vj * AttS[h * 256 + 128 + cwi]);
      hjv[r] = (f16)vj;
    }
    // C-fragment-major persist: one contiguous 16B store per tile
    *(v8h*)(hj_out + ((((size_t)blockIdx.x * 32 + tile) * 32 + lane) << 3)) = hjv;
  }
  __syncthreads();
  if (tid < 16 * NH) alpha_raw[(size_t)e0 * NH + tid] = siluf_(AlphaAcc[tid]);
}

// ---------------------------------------------------------------------------
__global__ void k_alpha_stats(const float* __restrict__ a, int E, float* __restrict__ stats) {
  __shared__ float ls[2 * NH];
  if (threadIdx.x < 2 * NH) ls[threadIdx.x] = 0.0f;
  __syncthreads();
  int n = E * NH;
  for (int i = blockIdx.x * blockDim.x + threadIdx.x; i < n; i += gridDim.x * blockDim.x) {
    float v = a[i];
    int   h = i & (NH - 1);
    atomicAdd(&ls[h], v);
    atomicAdd(&ls[NH + h], v * v);
  }
  __syncthreads();
  if (threadIdx.x < 2 * NH) atomicAdd(&stats[threadIdx.x], ls[threadIdx.x]);
}

__global__ void k_alpha_bn_segmax(const float* __restrict__ araw, int E,
                                  const float* __restrict__ stats,
                                  const float* __restrict__ g, const float* __restrict__ b,
                                  const int* __restrict__ eidx,
                                  float* __restrict__ aw, unsigned* __restrict__ segenc) {
  int i = blockIdx.x * blockDim.x + threadIdx.x;
  if (i >= E * NH) return;
  int   e = i >> 2, h = i & 3;
  float invE = 1.0f / (float)E;
  float mu   = stats[h] * invE;
  float var  = stats[NH + h] * invE - mu * mu;
  float an   = (araw[i] - mu) * rsqrtf(var + 1e-5f) * g[h] + b[h];
  float s    = siluf_(an);
  aw[i] = s;
  int node = eidx[e];
  atomicMax(&segenc[(size_t)node * NH + h], encf_(s));
}

__global__ void k_exp_segsum(float* __restrict__ aw, int E, const int* __restrict__ eidx,
                             const unsigned* __restrict__ segenc, float* __restrict__ segsum) {
  int i = blockIdx.x * blockDim.x + threadIdx.x;
  if (i >= E * NH) return;
  int   e = i >> 2, h = i & 3;
  int   node = eidx[e];
  float amax = decf_(segenc[(size_t)node * NH + h]);
  if (!(fabsf(amax) < 3.0e38f)) amax = 0.0f;  // NaN/Inf (empty segment) -> 0
  float ev = __expf(aw[i] - amax);
  aw[i] = ev;
  atomicAdd(&segsum[(size_t)node * NH + h], ev);
}

__global__ void k_node_init(const float* __restrict__ x, const float* __restrict__ bias,
                            float* __restrict__ node, int N) {
  int i = blockIdx.x * blockDim.x + threadIdx.x;
  if (i < N * DIM) node[i] = x[i] + bias[i & (DIM - 1)];
}

// hj is C-fragment-major: idx = ((blk*32 + tile)*32 + lane)*8 + r
__global__ __launch_bounds__(128) void k_aggregate(
    float* __restrict__ node, const f16* __restrict__ hj, const float* __restrict__ aw,
    const float* __restrict__ segsum, const int* __restrict__ eidx, int E) {
  __shared__ float w4[NH];
  __shared__ int   nid;
  int e = blockIdx.x;
  if (threadIdx.x == 0) nid = eidx[e];
  __syncthreads();
  int i = nid;
  if (threadIdx.x < NH)
    w4[threadIdx.x] = aw[(size_t)e * NH + threadIdx.x] /
                      (segsum[(size_t)i * NH + threadIdx.x] + 1e-16f);
  __syncthreads();
  int d   = threadIdx.x;
  int blk = e >> 4, m = e & 15, r = m & 7, lh = (m >> 3) << 4;
  float acc = 0.0f;
#pragma unroll
  for (int h = 0; h < NH; ++h) {
    int col  = h * DIM + d;
    int tile = col >> 4;
    int lane = (col & 15) + lh;
    acc += w4[h] * (float)hj[((((size_t)blk * 32 + tile) * 32 + lane) << 3) + r];
  }
  atomicAdd(&node[(size_t)i * DIM + d], 0.25f * acc);
}

// ---------------------------------------------------------------------------
// K7: m = sigmoid([n_i|n_j|ea]@lfw + lfb) * softplus([n_i|n_j|ea]@lsw + lsb)
//     (16x384 @ 384x128 WMMA, x2 weight matrices sharing A fragments)
__global__ __launch_bounds__(128) void k_edge_gemm(
    const float* __restrict__ node, const float* __restrict__ ea,
    const int* __restrict__ eidx, int E,
    const f16* __restrict__ lfp, const float* __restrict__ lfb,
    const f16* __restrict__ lsp, const float* __restrict__ lsb,
    float* __restrict__ mbuf, float* __restrict__ estats) {
  __shared__ __align__(32) f16 Az[16 * 384];   // 12 ksteps x 32 lanes x 16
  __shared__ float fb[DIM], sb[DIM];
  __shared__ float bsum[DIM], bsq[DIM];
  __shared__ int   si[16], sj[16];

  int tid = threadIdx.x;
  int e0  = blockIdx.x * 16;
  if (tid < 16) { si[tid] = eidx[e0 + tid]; sj[tid] = eidx[E + e0 + tid]; }
  if (tid < DIM) { fb[tid] = lfb[tid]; sb[tid] = lsb[tid]; bsum[tid] = 0.0f; bsq[tid] = 0.0f; }
  __syncthreads();
  for (int di = tid; di < 16 * 384; di += 128) {
    int e  = di & 15;
    int ln = (di >> 4) & 31;
    int ks = di >> 9;
    int r  = ln & 15;
    int c  = ks * 32 + a_klocal_(ln, e);
    float v;
    if (c < DIM)            v = node[(size_t)si[r] * DIM + c];
    else if (c < 2 * DIM)   v = node[(size_t)sj[r] * DIM + (c - DIM)];
    else                    v = ea[(size_t)(e0 + r) * DIM + (c - 2 * DIM)];
    Az[di] = (f16)v;
  }
  __syncthreads();

  int wave = tid >> 5, lane = tid & 31;
  int rowc = (lane & 16) ? 8 : 0;
  int colc = lane & 15;

#pragma unroll 1
  for (int t2 = 0; t2 < 2; ++t2) {
    int tile = wave + t2 * 4;  // 8 tiles -> 128 cols, NT = 8
    int n0   = tile * 16;
    v8f cf = {}, cs = {};
#pragma unroll
    for (int ks = 0; ks < 12; ++ks) {
      v16h a  = fragA(Az, ks, lane);
      v16h bf = fragB(lfp, ks * 8 + tile, lane);
      cf = WMMA_F32_F16(a, bf, cf);
      v16h bs = fragB(lsp, ks * 8 + tile, lane);
      cs = WMMA_F32_F16(a, bs, cs);
    }
    int col = n0 + colc;
#pragma unroll
    for (int r = 0; r < 8; ++r) {
      int   m = r + rowc;
      float v = sigmoidf_(cf[r] + fb[col]) * softplusf_(cs[r] + sb[col]);
      mbuf[(size_t)(e0 + m) * DIM + col] = v;
      atomicAdd(&bsum[col], v);
      atomicAdd(&bsq[col], v * v);
    }
  }
  __syncthreads();
  if (tid < DIM) {
    atomicAdd(&estats[tid], bsum[tid]);
    atomicAdd(&estats[DIM + tid], bsq[tid]);
  }
}

__global__ void k_edge_finalize(const float* __restrict__ ea, const float* __restrict__ mbuf,
                                const float* __restrict__ estats,
                                const float* __restrict__ g, const float* __restrict__ b,
                                float* __restrict__ out, int E) {
  int i = blockIdx.x * blockDim.x + threadIdx.x;
  if (i >= E * DIM) return;
  int   d    = i & (DIM - 1);
  float invE = 1.0f / (float)E;
  float mu   = estats[d] * invE;
  float var  = estats[DIM + d] * invE - mu * mu;
  out[i] = ea[i] + (mbuf[i] - mu) * rsqrtf(var + 1e-5f) * g[d] + b[d];
}

// ---------------------------------------------------------------------------
struct GcaoW {
  const float *att, *bias, *bng, *bnb, *lfb, *lsb, *beg, *beb;
};

static void run_gcao(const float* x, int N, const float* ea, const int* eidx, int E,
                     const GcaoW& w, const f16* Wp, const f16* lfp, const f16* lsp,
                     float* node_out, float* edge_out,
                     f16* hj, float* alpha_raw, float* alpha_w,
                     unsigned* segenc, float* segsum,
                     float* astats, float* estats, float* mbuf, hipStream_t stream) {
  hipMemsetAsync(astats, 0, 2 * NH * sizeof(float), stream);
  hipMemsetAsync(estats, 0, 2 * DIM * sizeof(float), stream);
  hipMemsetAsync(segenc, 0, (size_t)N * NH * sizeof(unsigned), stream);
  hipMemsetAsync(segsum, 0, (size_t)N * NH * sizeof(float), stream);

  k_gemm_h_alpha<<<E / 16, 128, 0, stream>>>(x, ea, eidx, E, Wp, w.att, hj, alpha_raw);
  k_alpha_stats<<<256, 256, 0, stream>>>(alpha_raw, E, astats);
  int n1 = E * NH;
  k_alpha_bn_segmax<<<(n1 + 255) / 256, 256, 0, stream>>>(alpha_raw, E, astats, w.bng, w.bnb,
                                                          eidx, alpha_w, segenc);
  k_exp_segsum<<<(n1 + 255) / 256, 256, 0, stream>>>(alpha_w, E, eidx, segenc, segsum);
  int n2 = N * DIM;
  k_node_init<<<(n2 + 255) / 256, 256, 0, stream>>>(x, w.bias, node_out, N);
  k_aggregate<<<E, 128, 0, stream>>>(node_out, hj, alpha_w, segsum, eidx, E);
  k_edge_gemm<<<E / 16, 128, 0, stream>>>(node_out, ea, eidx, E, lfp, w.lfb, lsp, w.lsb,
                                          mbuf, estats);
  k_edge_finalize<<<(E * DIM + 255) / 256, 256, 0, stream>>>(ea, mbuf, estats, w.beg, w.beb,
                                                             edge_out, E);
}

static size_t align_up_(size_t x) { return (x + 255) & ~(size_t)255; }

extern "C" void kernel_launch(void* const* d_in, const int* in_sizes, int n_in,
                              void* d_out, int out_size, void* d_ws, size_t ws_size,
                              hipStream_t stream) {
  (void)in_sizes; (void)n_in; (void)out_size; (void)ws_size;
  const int N_ATOM = 20000, N_BOND = 60000, N_ANGLE = 120000;

  const float* atom_feats = (const float*)d_in[0];
  const float* bond_attr  = (const float*)d_in[1];
  const float* trip_feats = (const float*)d_in[2];
  const int*   angle_idx  = (const int*)d_in[3];
  const int*   edge_idx   = (const int*)d_in[4];

  const float* W_a   = (const float*)d_in[5];
  const float* lfw_a = (const float*)d_in[10];
  const float* lsw_a = (const float*)d_in[12];
  GcaoW wa = { (const float*)d_in[6], (const float*)d_in[7], (const float*)d_in[8],
               (const float*)d_in[9], (const float*)d_in[11], (const float*)d_in[13],
               (const float*)d_in[14], (const float*)d_in[15] };

  const float* W_b   = (const float*)d_in[16];
  const float* lfw_b = (const float*)d_in[21];
  const float* lsw_b = (const float*)d_in[23];
  GcaoW wb = { (const float*)d_in[17], (const float*)d_in[18], (const float*)d_in[19],
               (const float*)d_in[20], (const float*)d_in[22], (const float*)d_in[24],
               (const float*)d_in[25], (const float*)d_in[26] };

  // ---- workspace layout ----
  char*  ws = (char*)d_ws;
  size_t off = 0;
  f16* aWp  = (f16*)(ws + off); off = align_up_(off + 256 * 512 * sizeof(f16));
  f16* bWp  = (f16*)(ws + off); off = align_up_(off + 256 * 512 * sizeof(f16));
  f16* alfp = (f16*)(ws + off); off = align_up_(off + 384 * 128 * sizeof(f16));
  f16* alsp = (f16*)(ws + off); off = align_up_(off + 384 * 128 * sizeof(f16));
  f16* blfp = (f16*)(ws + off); off = align_up_(off + 384 * 128 * sizeof(f16));
  f16* blsp = (f16*)(ws + off); off = align_up_(off + 384 * 128 * sizeof(f16));
  float* astats = (float*)(ws + off); off = align_up_(off + 2 * NH * sizeof(float));
  float* estats = (float*)(ws + off); off = align_up_(off + 2 * DIM * sizeof(float));
  float* alpha_raw = (float*)(ws + off); off = align_up_(off + (size_t)N_ANGLE * NH * sizeof(float));
  float* alpha_w   = (float*)(ws + off); off = align_up_(off + (size_t)N_ANGLE * NH * sizeof(float));
  unsigned* segenc = (unsigned*)(ws + off); off = align_up_(off + (size_t)N_BOND * NH * sizeof(unsigned));
  float* segsum    = (float*)(ws + off); off = align_up_(off + (size_t)N_BOND * NH * sizeof(float));
  float* bond_buf  = (float*)(ws + off); off = align_up_(off + (size_t)N_BOND * DIM * sizeof(float));
  f16* hj = (f16*)(ws + off);  // 120000*512*2 bytes; mbuf (120000*128*4) aliases it
  float* mbuf = (float*)(ws + off);

  // ---- pack weights into fragment-major f16 ----
  int nW = 256 * 512, nL = 384 * 128;
  k_pack_B<<<(nW + 255) / 256, 256, 0, stream>>>(W_a, aWp, 256, 512);
  k_pack_B<<<(nW + 255) / 256, 256, 0, stream>>>(W_b, bWp, 256, 512);
  k_pack_B<<<(nL + 255) / 256, 256, 0, stream>>>(lfw_a, alfp, 384, 128);
  k_pack_B<<<(nL + 255) / 256, 256, 0, stream>>>(lsw_a, alsp, 384, 128);
  k_pack_B<<<(nL + 255) / 256, 256, 0, stream>>>(lfw_b, blfp, 384, 128);
  k_pack_B<<<(nL + 255) / 256, 256, 0, stream>>>(lsw_b, blsp, 384, 128);

  // ---- outputs: (atom, bond2, trip) concatenated ----
  float* out       = (float*)d_out;
  float* atom_out  = out;
  float* bond2_out = out + (size_t)N_ATOM * DIM;
  float* trip_out  = out + (size_t)(N_ATOM + N_BOND) * DIM;

  // GCAO 1: line graph (nodes = bonds, edges = angles)
  run_gcao(bond_attr, N_BOND, trip_feats, angle_idx, N_ANGLE, wa, aWp, alfp, alsp,
           bond_buf, trip_out, hj, alpha_raw, alpha_w, segenc, segsum, astats, estats,
           mbuf, stream);

  // GCAO 2: atom graph (edge features = updated bonds)
  run_gcao(atom_feats, N_ATOM, bond_buf, edge_idx, N_BOND, wb, bWp, blfp, blsp,
           atom_out, bond2_out, hj, alpha_raw, alpha_w, segenc, segsum, astats, estats,
           mbuf, stream);
}
